// GNNDecoder_68143951118640
// MI455X (gfx1250) — matrix operand order repaired and checked by reading, not compiled, not run on hardware
//
#include <hip/hip_runtime.h>

typedef __attribute__((ext_vector_type(16))) _Float16 v16h;
typedef __attribute__((ext_vector_type(8)))  _Float16 v8h;
typedef __attribute__((ext_vector_type(8)))  float    v8f;
typedef __attribute__((ext_vector_type(4)))  float    v4f;

#define NNODES   2024
#define HID      128
#define BATCH    128
#define NROWS    (BATCH * NNODES)      // 259072 flattened (b,n) rows
#define ASTRIDE  264                   // padded row stride (halfs) for sA / sW
#define OSTRIDE  132                   // padded row stride (floats) for sOut
#define SA_BYTE_BASE (HID * ASTRIDE * 2)  // LDS byte offset of sA (dyn LDS @ 0)

// ---------------------------------------------------------------------------
// One GNN layer, fused: stencil-mean agg -> [128 rows x 256] @ [256 x 128]
// WMMA f16 GEMM -> +bias -> ReLU -> LayerNorm -> f16 out.
// Layer 0 additionally fuses the scalar embed: h0 = x*We+be, so agg0 is
// computed from the stencil mean of x directly (x0 != nullptr selects this).
// Layers 1..3 copy the "self" half of the tile with async global->LDS DMA.
// Block: 256 threads (8 waves), owns 128 consecutive flattened rows.
// Dynamic LDS: sW (128x264 h) + sA (128x264 h, reused as 128x132 f32) = 132KB.
// ---------------------------------------------------------------------------
__global__ __launch_bounds__(256)
void gnn_layer_kernel(const _Float16* __restrict__ hin,   // null for layer 0
                      const float* __restrict__ x0,       // non-null layer 0
                      const float* __restrict__ We,       // [128] embed weight
                      const float* __restrict__ be,       // [128] embed bias
                      _Float16* __restrict__ hout,
                      const float* __restrict__ W,        // [256*128] row-major
                      const float* __restrict__ bias,     // [128]
                      const float* __restrict__ lng,      // [128]
                      const float* __restrict__ lnb) {    // [128]
  extern __shared__ char dynsmem[];
  _Float16* sW = (_Float16*)dynsmem;                       // sW[o*ASTRIDE + k]
  _Float16* sA = (_Float16*)(dynsmem + SA_BYTE_BASE);      // sA[r*ASTRIDE + k]
  float*    sOut = (float*)sA;                             // reuse, same bytes/row

  const int tid = threadIdx.x;

  // ---- stage weights: transpose + f32->f16 ----
  for (int i = tid; i < 256 * HID; i += 256) {
    int k = i >> 7, o = i & 127;
    sW[o * ASTRIDE + k] = (_Float16)W[i];
  }

  // ---- build combined tile: [h | mean(neighbors)] in f16 ----
  {
    const int r = tid >> 1, half = tid & 1;
    const int R = blockIdx.x * 128 + r;
    const int n = R % NNODES;
    const bool hm1  = (n >= 1),  hp1  = (n <= NNODES - 2);
    const bool hm45 = (n >= 45), hp45 = (n <= NNODES - 46);
    const int cnt = (int)hm1 + (int)hp1 + (int)hm45 + (int)hp45;
    const float inv = 1.0f / (float)cnt;

    if (x0) {
      // -------- layer 0: embed fused, everything from scalar x --------
      if (half == 0) {
        const float xv = x0[R];
#pragma unroll
        for (int c = 0; c < HID; c += 8) {
          v8h o;
#pragma unroll
          for (int i = 0; i < 8; ++i)
            o[i] = (_Float16)(xv * We[c + i] + be[c + i]);
          *(v8h*)(sA + r * ASTRIDE + c) = o;
        }
      } else {
        float xs = 0.0f;
        if (hm1)  xs += x0[R - 1];
        if (hp1)  xs += x0[R + 1];
        if (hm45) xs += x0[R - 45];
        if (hp45) xs += x0[R + 45];
        const float xbar = xs * inv;
#pragma unroll
        for (int c = 0; c < HID; c += 8) {
          v8h o;
#pragma unroll
          for (int i = 0; i < 8; ++i)
            o[i] = (_Float16)(xbar * We[c + i] + be[c + i]);
          *(v8h*)(sA + r * ASTRIDE + HID + c) = o;
        }
      }
    } else {
      // -------- layers 1..3: read previous f16 activations --------
      const _Float16* self = hin + (size_t)R * HID;
      if (half == 0) {
        // async DMA: global -> LDS, 16B per op, no VGPR staging (ASYNCcnt)
        const unsigned lds_row = SA_BYTE_BASE + (unsigned)(r * ASTRIDE) * 2u;
#pragma unroll
        for (int c = 0; c < HID; c += 8) {
          unsigned lo = lds_row + (unsigned)c * 2u;
          const _Float16* gp = self + c;
          asm volatile("global_load_async_to_lds_b128 %0, %1, off"
                       :: "v"(lo), "v"(gp) : "memory");
        }
      } else {
        const _Float16 invh = (_Float16)inv;
#pragma unroll 4
        for (int c = 0; c < HID; c += 8) {
          v8h a;
#pragma unroll
          for (int i = 0; i < 8; ++i) a[i] = (_Float16)0.0f;
          if (hm1)  a = a + *(const v8h*)(self - HID + c);       // v_pk_add_f16
          if (hp1)  a = a + *(const v8h*)(self + HID + c);
          if (hm45) a = a + *(const v8h*)(self - 45 * HID + c);
          if (hp45) a = a + *(const v8h*)(self + 45 * HID + c);
          v8h o;
#pragma unroll
          for (int i = 0; i < 8; ++i) o[i] = a[i] * invh;        // v_pk_mul_f16
          *(v8h*)(sA + r * ASTRIDE + HID + c) = o;
        }
      }
    }
  }
  // drain this wave's async LDS writes before the workgroup barrier
  asm volatile("s_wait_asynccnt 0" ::: "memory");
  __syncthreads();

  // ---- WMMA GEMM: wave w -> rows [16w,16w+16), all 128 output cols ----
  const int wave = tid >> 5, lane = tid & 31;
  const int m = lane & 15, g = lane >> 4;

  float bl[8];
#pragma unroll
  for (int t = 0; t < 8; ++t) bl[t] = bias[t * 16 + m];

  v8f acc[8] = {};
  const _Float16* aRow = sA + (wave * 16 + m) * ASTRIDE;
#pragma unroll
  for (int kk = 0; kk < 8; ++kk) {
    const int ka = kk * 32 + g * 16;
    v8h alo = *(const v8h*)(aRow + ka);
    v8h ahi = *(const v8h*)(aRow + ka + 8);
    v16h afrag = __builtin_shufflevector(alo, ahi,
        0, 1, 2, 3, 4, 5, 6, 7, 8, 9, 10, 11, 12, 13, 14, 15);
#pragma unroll
    for (int t = 0; t < 8; ++t) {
      const _Float16* bRow = sW + (t * 16 + m) * ASTRIDE + kk * 32 + g * 16;
      v8h blo = *(const v8h*)(bRow);
      v8h bhi = *(const v8h*)(bRow + 8);
      v16h bfrag = __builtin_shufflevector(blo, bhi,
          0, 1, 2, 3, 4, 5, 6, 7, 8, 9, 10, 11, 12, 13, 14, 15);
      acc[t] = __builtin_amdgcn_wmma_f32_16x16x32_f16(
          false, afrag, false, bfrag, (short)0, acc[t], false, false);
    }
  }

  // ---- epilogue: bias + ReLU into f32 LDS (wave reuses its own sA rows) ----
#pragma unroll
  for (int t = 0; t < 8; ++t)
#pragma unroll
    for (int v = 0; v < 8; ++v) {
      int row = wave * 16 + g * 8 + v;
      float xv = fmaxf(acc[t][v] + bl[t], 0.0f);
      sOut[row * OSTRIDE + t * 16 + m] = xv;
    }
  __syncthreads();

  // ---- LayerNorm: thread pair per row (adjacent lanes), f16 store ----
  {
    const int r = tid >> 1, half = tid & 1;
    const float* rowp = sOut + r * OSTRIDE + half * 64;
    float s = 0.0f, sq = 0.0f;
#pragma unroll
    for (int c = 0; c < 64; c += 4) {
      v4f xv = *(const v4f*)(rowp + c);
#pragma unroll
      for (int i = 0; i < 4; ++i) { s += xv[i]; sq += xv[i] * xv[i]; }
    }
    s  += __shfl_xor(s, 1, 32);
    sq += __shfl_xor(sq, 1, 32);
    float mu = s * (1.0f / HID);
    float var = sq * (1.0f / HID) - mu * mu;
    float rs = rsqrtf(var + 1e-5f);
    size_t R = (size_t)blockIdx.x * 128 + r;
    _Float16* outp = hout + R * HID + half * 64;
    const float* gg = lng + half * 64;
    const float* bb = lnb + half * 64;
#pragma unroll
    for (int c = 0; c < 64; c += 8) {
      v8h o;
#pragma unroll
      for (int i = 0; i < 8; ++i)
        o[i] = (_Float16)((rowp[c + i] - mu) * rs * gg[c + i] + bb[c + i]);
      *(v8h*)(outp + c) = o;
    }
  }
}

// ---------------------------------------------------------------------------
// graph[b, c] = mean_n h[b, n, c]
// ---------------------------------------------------------------------------
__global__ void gnn_reduce_kernel(const _Float16* __restrict__ h,
                                  float* __restrict__ graph) {
  __shared__ float part[128];
  int b = blockIdx.x;
  int c = threadIdx.x & 127, half = threadIdx.x >> 7;
  const _Float16* base = h + ((size_t)b * NNODES + half * 1012) * HID + c;
  float s = 0.0f;
  for (int n = 0; n < 1012; ++n) s += (float)base[(size_t)n * HID];
  if (half) part[c] = s;
  __syncthreads();
  if (!half) graph[b * HID + c] = (s + part[c]) * (1.0f / (float)NNODES);
}

// ---------------------------------------------------------------------------
// hid = relu(graph @ W1 + b1); out = hid @ W2 + b2
// ---------------------------------------------------------------------------
__global__ void gnn_mlp_kernel(const float* __restrict__ graph,
                               const float* __restrict__ W1,
                               const float* __restrict__ b1,
                               const float* __restrict__ W2,
                               const float* __restrict__ b2,
                               float* __restrict__ out) {
  extern __shared__ char dynsmem[];
  float* sHid = (float*)dynsmem;   // [128][128]
  int tid = threadIdx.x;
  for (int i = tid; i < BATCH * HID; i += 256) {
    int b = i >> 7, j = i & 127;
    float s = b1[j];
    for (int k = 0; k < HID; ++k) s += graph[b * HID + k] * W1[k * HID + j];
    sHid[i] = fmaxf(s, 0.0f);
  }
  __syncthreads();
  int b = tid >> 1, o = tid & 1;
  float s = b2[o];
  for (int j = 0; j < HID; ++j) s += sHid[b * HID + j] * W2[j * 2 + o];
  out[b * 2 + o] = s;
}

// ---------------------------------------------------------------------------
extern "C" void kernel_launch(void* const* d_in, const int* in_sizes, int n_in,
                              void* d_out, int out_size, void* d_ws, size_t ws_size,
                              hipStream_t stream) {
  const float* x  = (const float*)d_in[0];
  const float* We = (const float*)d_in[1];
  const float* be = (const float*)d_in[2];
  const float* gW = (const float*)d_in[3];   // [4][256][128]
  const float* gb = (const float*)d_in[4];   // [4][128]
  const float* lg = (const float*)d_in[5];   // [4][128]
  const float* lb = (const float*)d_in[6];   // [4][128]
  const float* W1 = (const float*)d_in[7];
  const float* b1 = (const float*)d_in[8];
  const float* W2 = (const float*)d_in[9];
  const float* b2 = (const float*)d_in[10];
  float* out = (float*)d_out;

  const size_t hbytes = (size_t)NROWS * HID * sizeof(_Float16); // 66.3 MB
  _Float16* hA = (_Float16*)d_ws;
  _Float16* hB = (_Float16*)((char*)d_ws + hbytes);
  float* graph = (float*)((char*)d_ws + 2 * hbytes);

  const size_t lds = (size_t)2 * HID * ASTRIDE * sizeof(_Float16); // 132 KB

  // layer 0: embed fused (reads x), writes hA
  gnn_layer_kernel<<<NROWS / 128, 256, lds, stream>>>(
      nullptr, x, We, be, hA,
      gW, gb, lg, lb);
  // layers 1..3: ping-pong hA -> hB -> hA -> hB
  _Float16* hin = hA; _Float16* hout = hB;
  for (int l = 1; l < 4; ++l) {
    gnn_layer_kernel<<<NROWS / 128, 256, lds, stream>>>(
        hin, nullptr, nullptr, nullptr, hout,
        gW + (size_t)l * 256 * HID, gb + l * HID, lg + l * HID, lb + l * HID);
    _Float16* t = hin; hin = hout; hout = t;
  }
  // final activations are in hB (== hin after 3 swaps)

  gnn_reduce_kernel<<<BATCH, 256, 0, stream>>>(hin, graph);
  gnn_mlp_kernel<<<1, 256, BATCH * HID * sizeof(float), stream>>>(
      graph, W1, b1, W2, b2, out);
}
